// GCN_82179904241990
// MI455X (gfx1250) — matrix operand order repaired and checked by reading, not compiled, not run on hardware
//
#include <hip/hip_runtime.h>
#include <hip/hip_bf16.h>

#define N_NODES 100000
#define NFEAT   512
#define NHID    256
#define NCLASS  41
#define NCLS_PAD 48

typedef __attribute__((ext_vector_type(16))) __bf16       v16bf;
typedef __attribute__((ext_vector_type(8)))  float        v8f;
typedef __attribute__((ext_vector_type(8)))  unsigned int v8u;
typedef int ivec4 __attribute__((vector_size(16)));        // matches builtin proto

union Frag { v8u u; v16bf bf; };

#if __has_builtin(__builtin_amdgcn_global_load_async_to_lds_b128) && \
    __has_builtin(__builtin_amdgcn_s_wait_asynccnt)
#define HAVE_ASYNC_LDS 1
#else
#define HAVE_ASYNC_LDS 0
#endif

// copy 16 bytes global -> LDS (async path when available)
__device__ __forceinline__ void stage_b128(const unsigned int* g, unsigned int* l) {
#if HAVE_ASYNC_LDS
  __builtin_amdgcn_global_load_async_to_lds_b128(
      (__attribute__((address_space(1))) ivec4*)(unsigned long long)g,
      (__attribute__((address_space(3))) ivec4*)(unsigned int)(unsigned long long)l,
      0, 0);
#else
  *(uint4*)l = *(const uint4*)g;
#endif
}

__device__ __forceinline__ void stage_fence() {
#if HAVE_ASYNC_LDS
  __builtin_amdgcn_s_wait_asynccnt(0);
#endif
  __syncthreads();
}

// round-to-nearest-even f32 -> bf16 pair packed in a dword (cold pack kernels)
__device__ __forceinline__ unsigned int bf16pack2(float a, float b) {
  unsigned int ua = __float_as_uint(a);
  unsigned int ub = __float_as_uint(b);
  ua = (ua + 0x7FFFu + ((ua >> 16) & 1u)) >> 16;
  ub = (ub + 0x7FFFu + ((ub >> 16) & 1u)) >> 16;
  return (ua & 0xFFFFu) | (ub << 16);
}

// ---------------------------------------------------------------------------
// Pack W1 [512,256] f32 into bf16 B-fragment layout:
//   index = ((kk*NHID + col)*2 + kh)*8 + d  holds bf16 pair for
//   K = kk*32 + (d>>2)*16 + kh*8 + (d&3)*2  (and K+1), column col.
// ---------------------------------------------------------------------------
__global__ void __launch_bounds__(256) pack_w1_kernel(const float* __restrict__ w1,
                                                      unsigned int* __restrict__ w1p) {
  const int idx = blockIdx.x * blockDim.x + threadIdx.x;     // 65536 entries
  const int d  = idx & 7;
  const int kh = (idx >> 3) & 1;
  const int c  = (idx >> 4) & (NHID - 1);
  const int kk = idx >> 12;
  const int K  = kk * 32 + ((d >> 2) << 4) + (kh << 3) + ((d & 3) << 1);
  w1p[idx] = bf16pack2(w1[(size_t)K * NHID + c], w1[(size_t)(K + 1) * NHID + c]);
}

// W2 [256,41] padded to 48 columns, same fragment layout
__global__ void __launch_bounds__(256) pack_w2_kernel(const float* __restrict__ w2,
                                                      unsigned int* __restrict__ w2p) {
  const int idx = blockIdx.x * blockDim.x + threadIdx.x;
  if (idx >= (NHID / 32) * NCLS_PAD * 16) return;            // 6144 entries
  const int d  = idx & 7;
  const int kh = (idx >> 3) & 1;
  const int c  = (idx >> 4) % NCLS_PAD;
  const int kk = idx / (NCLS_PAD * 16);
  const int K  = kk * 32 + ((d >> 2) << 4) + (kh << 3) + ((d & 3) << 1);
  const float a = (c < NCLASS) ? w2[(size_t)K * NCLASS + c] : 0.0f;
  const float b = (c < NCLASS) ? w2[(size_t)(K + 1) * NCLASS + c] : 0.0f;
  w2p[idx] = bf16pack2(a, b);
}

// ---------------------------------------------------------------------------
// rows is sorted -> CSR row pointer (N+1 entries)
// ---------------------------------------------------------------------------
__global__ void __launch_bounds__(256) build_rowptr_kernel(const int* __restrict__ rows,
                                                           int* __restrict__ rowptr, int nnz) {
  const int e = blockIdx.x * blockDim.x + threadIdx.x;
  if (e >= nnz) return;
  const int r  = rows[e];
  const int rp = (e == 0) ? -1 : rows[e - 1];
  for (int q = rp + 1; q <= r; ++q) rowptr[q] = e;
  if (e == nnz - 1)
    for (int q = r + 1; q <= N_NODES; ++q) rowptr[q] = nnz;
}

// ---------------------------------------------------------------------------
// GEMM1: XW = X @ W1. One wave owns a 16-row x 256-col strip (16 accumulators,
// A fragment loaded once per K-step -> 16 WMMAs). The shared packed-W1 K-tile
// (16KB) is staged into LDS per block, double-buffered, via async global->LDS;
// inner loop reads B fragments with ds_load_b128 instead of waiting on L2.
// ---------------------------------------------------------------------------
__global__ void __launch_bounds__(256) gemm1_wmma(const float* __restrict__ x,
                                                  const unsigned int* __restrict__ w1p,
                                                  float* __restrict__ xw) {
  __shared__ unsigned int sb[2][4096];                        // 2 x 16KB B tiles
  const int tid  = threadIdx.x;
  const int lane = tid & 31;
  const int tm   = blockIdx.x * 8 + (tid >> 5);               // 6250 strips
  const bool active = tm < N_NODES / 16;
  const int tmc  = active ? tm : (N_NODES / 16 - 1);          // clamp for safe addressing
  const int m    = lane & 15;                                 // A: M index / B & C: N index
  const int kh   = lane >> 4;                                 // lane half
  const float* xrow = x + (size_t)(tmc * 16 + m) * NFEAT;

  // stage K-tile 0
#pragma unroll
  for (int c = 0; c < 4; ++c)
    stage_b128(w1p + (size_t)(tid + c * 256) * 4, &sb[0][(tid + c * 256) * 4]);
  stage_fence();

  v8f acc[16];
#pragma unroll
  for (int t = 0; t < 16; ++t) acc[t] = (v8f){};

  for (int kk = 0; kk < NFEAT / 32; ++kk) {
    const int cur = kk & 1;
    if (kk + 1 < NFEAT / 32) {                                // stage next tile (async)
#pragma unroll
      for (int c = 0; c < 4; ++c)
        stage_b128(w1p + ((size_t)(kk + 1) << 12) + (size_t)(tid + c * 256) * 4,
                   &sb[cur ^ 1][(tid + c * 256) * 4]);
      __builtin_prefetch(xrow + (kk + 1) * 32, 0, 3);
    }

    Frag a;
#pragma unroll
    for (int d = 0; d < 8; ++d) {
      const int K = kk * 32 + ((d >> 2) << 4) + (kh << 3) + ((d & 3) << 1);
      const float2 t = *(const float2*)(xrow + K);
      a.bf[2 * d]     = (__bf16)t.x;                          // native v_cvt_pk_bf16_f32
      a.bf[2 * d + 1] = (__bf16)t.y;
    }

#pragma unroll
    for (int tn = 0; tn < 16; ++tn) {
      const unsigned int* bp = &sb[cur][tn * 256 + m * 16 + kh * 8];
      const uint4 bl = *(const uint4*)(bp);                   // ds_load_b128
      const uint4 bh = *(const uint4*)(bp + 4);
      Frag b;
      b.u[0] = bl.x; b.u[1] = bl.y; b.u[2] = bl.z; b.u[3] = bl.w;
      b.u[4] = bh.x; b.u[5] = bh.y; b.u[6] = bh.z; b.u[7] = bh.w;
      acc[tn] = __builtin_amdgcn_wmma_f32_16x16x32_bf16(false, a.bf, false, b.bf,
                                                        (short)0, acc[tn], false, false);
    }

    if (kk + 1 < NFEAT / 32) stage_fence();                   // next tile landed; all reads done
  }

  if (active) {
    float* outb = xw + (size_t)(tm * 16) * NHID;
#pragma unroll
    for (int tn = 0; tn < 16; ++tn)
#pragma unroll
      for (int v = 0; v < 8; ++v)
        outb[(size_t)(v + 8 * kh) * NHID + tn * 16 + m] = acc[tn][v];
  }
}

// ---------------------------------------------------------------------------
// SpMM (CSR, one wave per row): dst[row,:] = sum_e vals[e]*src[cols[e],:]
// 32 lanes x 2 float4 cover the 256-wide feature dim. Optional bias+ReLU.
// ---------------------------------------------------------------------------
__global__ void __launch_bounds__(256) spmm_row_kernel(const int* __restrict__ rowptr,
                                                       const int* __restrict__ cols,
                                                       const float* __restrict__ vals,
                                                       const float* __restrict__ src,
                                                       const float* __restrict__ bias,
                                                       float* __restrict__ dst,
                                                       int do_bias_relu) {
  const int lane = threadIdx.x & 31;
  const int row  = blockIdx.x * 8 + (threadIdx.x >> 5);
  if (row >= N_NODES) return;
  const int e0 = rowptr[row];
  const int e1 = rowptr[row + 1];

  float4 a0 = {0.f, 0.f, 0.f, 0.f};
  float4 a1 = {0.f, 0.f, 0.f, 0.f};
  for (int e = e0; e < e1; ++e) {
    const int   c = cols[e];
    const float v = vals[e];
    const float4* s = (const float4*)(src + (size_t)c * NHID);
    const float4 g0 = s[lane];
    const float4 g1 = s[lane + 32];
    a0.x = fmaf(g0.x, v, a0.x); a0.y = fmaf(g0.y, v, a0.y);
    a0.z = fmaf(g0.z, v, a0.z); a0.w = fmaf(g0.w, v, a0.w);
    a1.x = fmaf(g1.x, v, a1.x); a1.y = fmaf(g1.y, v, a1.y);
    a1.z = fmaf(g1.z, v, a1.z); a1.w = fmaf(g1.w, v, a1.w);
  }
  if (do_bias_relu) {
    const float4 b0 = ((const float4*)bias)[lane];
    const float4 b1 = ((const float4*)bias)[lane + 32];
    a0.x = fmaxf(a0.x + b0.x, 0.f); a0.y = fmaxf(a0.y + b0.y, 0.f);
    a0.z = fmaxf(a0.z + b0.z, 0.f); a0.w = fmaxf(a0.w + b0.w, 0.f);
    a1.x = fmaxf(a1.x + b1.x, 0.f); a1.y = fmaxf(a1.y + b1.y, 0.f);
    a1.z = fmaxf(a1.z + b1.z, 0.f); a1.w = fmaxf(a1.w + b1.w, 0.f);
  }
  float4* d = (float4*)(dst + (size_t)row * NHID);
  d[lane]      = a0;
  d[lane + 32] = a1;
}

// ---------------------------------------------------------------------------
// GEMM2: logits = AH @ W2 (padded to 48 cols). Whole packed W2 (24KB) staged
// to LDS once; one wave = 16 rows x 48 cols (3 WMMA tiles sharing A fragment).
// ---------------------------------------------------------------------------
__global__ void __launch_bounds__(256) gemm2_wmma(const float* __restrict__ ah,
                                                  const unsigned int* __restrict__ w2p,
                                                  float* __restrict__ logits) {
  __shared__ unsigned int s2[6144];                           // 24KB packed W2
  const int tid  = threadIdx.x;
#pragma unroll
  for (int c = 0; c < 6; ++c)
    stage_b128(w2p + (size_t)(tid + c * 256) * 4, &s2[(tid + c * 256) * 4]);
  stage_fence();

  const int lane = tid & 31;
  const int tm   = blockIdx.x * 8 + (tid >> 5);
  if (tm >= N_NODES / 16) return;
  const int m  = lane & 15;
  const int kh = lane >> 4;
  const float* arow = ah + (size_t)(tm * 16 + m) * NHID;

  v8f acc0 = {}, acc1 = {}, acc2 = {};
  for (int kk = 0; kk < NHID / 32; ++kk) {
    Frag a;
#pragma unroll
    for (int d = 0; d < 8; ++d) {
      const int K = kk * 32 + ((d >> 2) << 4) + (kh << 3) + ((d & 3) << 1);
      const float2 t = *(const float2*)(arow + K);
      a.bf[2 * d]     = (__bf16)t.x;
      a.bf[2 * d + 1] = (__bf16)t.y;
    }
#pragma unroll
    for (int t = 0; t < 3; ++t) {
      const unsigned int* bp = &s2[kk * 768 + (t * 16 + m) * 16 + kh * 8];
      const uint4 bl = *(const uint4*)(bp);                   // ds_load_b128
      const uint4 bh = *(const uint4*)(bp + 4);
      Frag b;
      b.u[0] = bl.x; b.u[1] = bl.y; b.u[2] = bl.z; b.u[3] = bl.w;
      b.u[4] = bh.x; b.u[5] = bh.y; b.u[6] = bh.z; b.u[7] = bh.w;
      if (t == 0)      acc0 = __builtin_amdgcn_wmma_f32_16x16x32_bf16(false, a.bf, false, b.bf, (short)0, acc0, false, false);
      else if (t == 1) acc1 = __builtin_amdgcn_wmma_f32_16x16x32_bf16(false, a.bf, false, b.bf, (short)0, acc1, false, false);
      else             acc2 = __builtin_amdgcn_wmma_f32_16x16x32_bf16(false, a.bf, false, b.bf, (short)0, acc2, false, false);
    }
  }

  float* outb = logits + (size_t)(tm * 16) * NCLS_PAD;
#pragma unroll
  for (int v = 0; v < 8; ++v) {
    const size_t ro = (size_t)(v + 8 * kh) * NCLS_PAD;
    outb[ro + m]      = acc0[v];
    outb[ro + 16 + m] = acc1[v];
    outb[ro + 32 + m] = acc2[v];
  }
}

// ---------------------------------------------------------------------------
// log_softmax over 41 classes, one wave per row (wave32 xor-shuffle reduce)
// ---------------------------------------------------------------------------
__global__ void __launch_bounds__(256) logsoftmax_kernel(const float* __restrict__ logits,
                                                         const float* __restrict__ b2,
                                                         float* __restrict__ out) {
  const int lane = threadIdx.x & 31;
  const int row  = blockIdx.x * 8 + (threadIdx.x >> 5);
  if (row >= N_NODES) return;
  const float* lrow = logits + (size_t)row * NCLS_PAD;

  const float v0 = lrow[lane] + b2[lane];                      // lanes 0..31 all < 41
  const bool  hi = (lane + 32) < NCLASS;
  const float v1 = hi ? (lrow[lane + 32] + b2[lane + 32]) : -3.0e38f;

  float mx = fmaxf(v0, v1);
#pragma unroll
  for (int off = 16; off > 0; off >>= 1) mx = fmaxf(mx, __shfl_xor(mx, off, 32));

  float s = __expf(v0 - mx) + (hi ? __expf(v1 - mx) : 0.0f);
#pragma unroll
  for (int off = 16; off > 0; off >>= 1) s += __shfl_xor(s, off, 32);

  const float lse = mx + __logf(s);
  float* orow = out + (size_t)row * NCLASS;
  orow[lane] = v0 - lse;
  if (hi) orow[lane + 32] = v1 - lse;
}

// ---------------------------------------------------------------------------
extern "C" void kernel_launch(void* const* d_in, const int* in_sizes, int n_in,
                              void* d_out, int out_size, void* d_ws, size_t ws_size,
                              hipStream_t stream) {
  const float* x        = (const float*)d_in[0];
  const int*   adj_rows = (const int*)  d_in[1];
  const int*   adj_cols = (const int*)  d_in[2];
  const float* adj_vals = (const float*)d_in[3];
  const float* w1       = (const float*)d_in[4];
  const float* b1       = (const float*)d_in[5];
  const float* w2       = (const float*)d_in[6];
  const float* b2       = (const float*)d_in[7];
  float* out = (float*)d_out;
  const int nnz = in_sizes[1];

  char* ws = (char*)d_ws;
  size_t off = 0;
  float* xw     = (float*)(ws + off); off += (size_t)N_NODES * NHID * 4;      // reused as AH
  float* h      = (float*)(ws + off); off += (size_t)N_NODES * NHID * 4;
  float* logits = (float*)(ws + off); off += (size_t)N_NODES * NCLS_PAD * 4;
  unsigned int* w1p = (unsigned int*)(ws + off); off += (size_t)(NFEAT / 32) * NHID * 16 * 4;
  unsigned int* w2p = (unsigned int*)(ws + off); off += (size_t)(NHID / 32) * NCLS_PAD * 16 * 4;
  int* rowptr   = (int*)(ws + off);   off += (size_t)(N_NODES + 1) * 4;

  pack_w1_kernel<<<(NFEAT / 32) * NHID * 16 / 256, 256, 0, stream>>>(w1, w1p);
  pack_w2_kernel<<<((NHID / 32) * NCLS_PAD * 16 + 255) / 256, 256, 0, stream>>>(w2, w2p);
  build_rowptr_kernel<<<(nnz + 255) / 256, 256, 0, stream>>>(adj_rows, rowptr, nnz);

  // XW = X @ W1  (WMMA bf16, LDS-staged B, X read once)
  gemm1_wmma<<<(N_NODES / 16 + 7) / 8, 256, 0, stream>>>(x, w1p, xw);
  // H = relu(A @ XW + b1)
  spmm_row_kernel<<<N_NODES / 8, 256, 0, stream>>>(rowptr, adj_cols, adj_vals, xw, b1, h, 1);
  // AH = A @ H  (reuse xw buffer)
  spmm_row_kernel<<<N_NODES / 8, 256, 0, stream>>>(rowptr, adj_cols, adj_vals, h, b1, xw, 0);
  // logits = AH @ W2 (padded, WMMA bf16, LDS-staged B)
  gemm2_wmma<<<(N_NODES / 16 + 7) / 8, 256, 0, stream>>>(xw, w2p, logits);
  // out = log_softmax(logits + b2)
  logsoftmax_kernel<<<N_NODES / 8, 256, 0, stream>>>(logits, b2, out);
}
